// RGBDClassifier_17068200034919
// MI455X (gfx1250) — matrix operand order repaired and checked by reading, compiled, not run on hardware
//
#include <hip/hip_runtime.h>
#include <math.h>
#include <stdint.h>

// ---------------------------------------------------------------------------
// CDNA5 (gfx1250) implementation of the RGBD 2.5D-conv classifier.
// Matrix math: v_wmma_f32_16x16x32_bf16 (fp32 accumulate).
// fc1 weight stream: Tensor Data Mover (tensor_load_to_lds) double-buffered.
// ---------------------------------------------------------------------------

typedef __bf16        v16bf __attribute__((ext_vector_type(16)));
typedef float         v8f   __attribute__((ext_vector_type(8)));
typedef float         v4f   __attribute__((ext_vector_type(4)));
typedef unsigned int  v4u   __attribute__((ext_vector_type(4)));
typedef int           v4i   __attribute__((ext_vector_type(4)));
typedef int           v8i   __attribute__((ext_vector_type(8)));

#define IMG_H 224
#define IMG_W 224
#define NB    16   // batch
#define OC    16   // output channels of both convs

#if defined(__has_builtin)
#if __has_builtin(__builtin_amdgcn_tensor_load_to_lds) && \
    __has_builtin(__builtin_amdgcn_s_wait_tensorcnt)
#define HAVE_TDM 1
#if __clang_major__ >= 23
#define TDM_SIX_ARGS 1
#endif
#endif
#endif

// low 32 bits of a generic pointer to a __shared__ object == LDS byte offset
__device__ __forceinline__ unsigned lds_off(const void* p)
{
    return (unsigned)(unsigned long long)(uintptr_t)p;
}

__device__ __forceinline__ void pack8(v16bf& d, int base, v4f x, v4f y)
{
    d[base + 0] = (__bf16)x[0]; d[base + 1] = (__bf16)x[1];
    d[base + 2] = (__bf16)x[2]; d[base + 3] = (__bf16)x[3];
    d[base + 4] = (__bf16)y[0]; d[base + 5] = (__bf16)y[1];
    d[base + 6] = (__bf16)y[2]; d[base + 7] = (__bf16)y[3];
}

#if defined(HAVE_TDM)
// TDM 2D tile load: 16 rows x 32 fp32 cols, row stride = ktot elements,
// into LDS at lds_byte_off. D# per ISA 08_async_tensor §8.3/8.4.
__device__ __forceinline__ void tdm_load_tile_16x32(const float* gptr,
                                                    unsigned lds_byte_off,
                                                    unsigned ktot)
{
    unsigned long long ga = (unsigned long long)(uintptr_t)gptr;
    v4u g0;
    g0[0] = 1u;                                           // count=1, no gather
    g0[1] = lds_byte_off;                                 // lds_addr
    g0[2] = (unsigned)(ga & 0xffffffffu);                 // global_addr[31:0]
    g0[3] = (unsigned)((ga >> 32) & 0x1ffffffu)           // global_addr[56:32]
          | (2u << 30);                                   // type=2 ("image")
    unsigned long long st0 = (unsigned long long)ktot;    // dim0 stride (elems)
    v8i g1;
    g1[0] = (int)(2u << 16);                              // data_size=4B; mask=0
    g1[1] = (int)((ktot & 0xffffu) << 16);                // tensor_dim0[15:0]
    g1[2] = (int)(((ktot >> 16) & 0xffffu)                // tensor_dim0[31:16]
          | (16u << 16));                                 // tensor_dim1=16
    g1[3] = (int)(32u << 16);                             // tile_dim0=32
    g1[4] = (int)16u;                                     // tile_dim1=16, dim2=0
    g1[5] = (int)(unsigned)(st0 & 0xffffffffu);           // dim0_stride lo
    g1[6] = (int)(unsigned)((st0 >> 32) & 0xffffu);       // dim0_stride hi
    g1[7] = 0;
    v4i gz = { 0, 0, 0, 0 };
#if defined(TDM_SIX_ARGS)
    v8i gz8 = { 0, 0, 0, 0, 0, 0, 0, 0 };
    __builtin_amdgcn_tensor_load_to_lds(g0, g1, gz, gz, gz8, 0);
#else
    __builtin_amdgcn_tensor_load_to_lds(g0, g1, gz, gz, 0);
#endif
}
#endif

// ---------------------------------------------------------------------------
// 2.5D conv as implicit GEMM:  D(O=16 x 16 pixels) = sum_s A_s(16xK) B_s(Kx16)
// K per set = C*9 padded to T*32; mask m_s multiplies B. One wave per tile.
// A lane layout (16-bit A 16x32): lane l: M=l%16; K=(v>=4?16:0)+(l/16)*8+2(v&3)+hh
// B (32x16): lane l: N=l%16; K=(v>=4?16:0)+(l/16)*8+2(v&3)+hh
// D (16x16 f32): VGPR r: M=r+8*(l/16), N=l%16
// ---------------------------------------------------------------------------
template <int C, int T>
__global__ __launch_bounds__(32) void conv25d_wmma_kernel(
    const float* __restrict__ in,     // [NB][C][H][W]
    const float* __restrict__ depth,  // [NB][1][H][W]
    const float* __restrict__ fxp,    // [1]
    const float* __restrict__ wt0,
    const float* __restrict__ wt1,
    const float* __restrict__ wt2,    // each [OC][C][3][3]
    const float* __restrict__ bias,   // [OC]
    float* __restrict__ out,          // [NB][OC][H][W], ReLU applied
    int H, int W)
{
    __shared__ float lin[C][3][18];   // input patch rows (w0-1 .. w0+16)
    __shared__ float ldep[3][18];     // depth patch rows
    __shared__ float lmsk[27][16];    // masks: [set*9 + tap][pixel]

    const int n     = blockIdx.z;
    const int h     = blockIdx.y;
    const int wbase = blockIdx.x * 16;
    const int tid   = threadIdx.x;
    const int jcol  = tid & 15;
    const int g     = tid >> 4;

    // ---- cooperative stage of input + depth patch rows into LDS ----
    for (int i = tid; i < C * 3 * 18; i += 32) {
        int c  = i / (3 * 18);
        int r  = (i / 18) % 3;
        int ww = i % 18;
        int gh = h + r - 1, gw = wbase + ww - 1;
        float v = 0.f;
        if (gh >= 0 && gh < H && gw >= 0 && gw < W)
            v = in[((n * C + c) * H + gh) * W + gw];
        lin[c][r][ww] = v;
    }
    for (int i = tid; i < 3 * 18; i += 32) {
        int r = i / 18, ww = i % 18;
        int gh = h + r - 1, gw = wbase + ww - 1;
        float v = 0.f;
        if (gh >= 0 && gh < H && gw >= 0 && gw < W)
            v = depth[(n * H + gh) * W + gw];
        ldep[r][ww] = v;
    }
    __syncthreads();

    // ---- per-pixel depth masks (exact reference semantics) ----
    if (tid < 16) {
        const float fx   = fxp[0];
        const float ctr  = ldep[1][tid + 1];
        const float c0   = (ctr != 0.f) ? ctr : 0.f;
        const float gr   = c0 / fx;
        const float half = 0.5f * gr;
        #pragma unroll
        for (int kk = 0; kk < 9; ++kk) {
            float dd = ldep[kk / 3][tid + (kk % 3)];
            float dv = (dd != 0.f && ctr != 0.f) ? dd : 0.f;
            lmsk[0 * 9 + kk][tid] = (fabsf(dv - (c0 + gr)) <= half) ? 1.f : 0.f;
            lmsk[1 * 9 + kk][tid] = (fabsf(dv - c0)        <  half) ? 1.f : 0.f;
            lmsk[2 * 9 + kk][tid] = (fabsf(dv - (c0 - gr)) <= half) ? 1.f : 0.f;
        }
    }
    __syncthreads();

    const float* wp[3] = { wt0, wt1, wt2 };
    v8f acc = {};

    #pragma unroll
    for (int s = 0; s < 3; ++s) {
        #pragma unroll
        for (int t = 0; t < T; ++t) {
            v16bf a, b;
            #pragma unroll
            for (int v = 0; v < 8; ++v) {
                #pragma unroll
                for (int hh = 0; hh < 2; ++hh) {
                    // A: weights (row = output channel jcol)
                    int Ka  = ((v >= 4) ? 16 : 0) + g * 8 + 2 * (v & 3) + hh;
                    int kla = t * 32 + Ka;
                    float av = (kla < C * 9) ? wp[s][jcol * C * 9 + kla] : 0.f;
                    a[2 * v + hh] = (__bf16)av;
                    // B: masked input patch (col = pixel jcol)
                    int Kb  = ((v >= 4) ? 16 : 0) + g * 8 + 2 * (v & 3) + hh;
                    int klb = t * 32 + Kb;
                    float bv = 0.f;
                    if (klb < C * 9) {
                        int c  = klb / 9;
                        int kk = klb % 9;
                        bv = lin[c][kk / 3][jcol + (kk % 3)] * lmsk[s * 9 + kk][jcol];
                    }
                    b[2 * v + hh] = (__bf16)bv;
                }
            }
            acc = __builtin_amdgcn_wmma_f32_16x16x32_bf16(
                false, a, false, b, (short)0, acc, false, false);
        }
    }

    // ---- bias + ReLU + store (D layout) ----
    #pragma unroll
    for (int r = 0; r < 8; ++r) {
        int o   = r + 8 * g;
        float v = acc[r] + bias[o];
        out[((n * OC + o) * H + h) * W + wbase + jcol] = (v > 0.f) ? v : 0.f;
    }
}

// ---------------------------------------------------------------------------
// 2x2 max-pool (inputs already ReLU'd)
// ---------------------------------------------------------------------------
__global__ void maxpool2_kernel(const float* __restrict__ x,
                                float* __restrict__ p, int total)
{
    int i = blockIdx.x * blockDim.x + threadIdx.x;
    if (i >= total) return;
    int xx = i % 112;
    int y  = (i / 112) % 112;
    int c  = (i / (112 * 112)) % OC;
    int n  = i / (112 * 112 * OC);
    const float* b = x + ((n * OC + c) * IMG_H + 2 * y) * IMG_W + 2 * xx;
    p[i] = fmaxf(fmaxf(b[0], b[1]), fmaxf(b[IMG_W], b[IMG_W + 1]));
}

// ---------------------------------------------------------------------------
// fc1: out1[16,128] += p[16,K] * fc1_w[128,K]^T   (split-K, fp32 atomics)
// grid.x = 8 (feature tiles of 16), grid.y = split-K index.
// TDM path: per K-step, DMA the 16x32 A (activations) and B (weights) tiles
// into LDS (double buffered, s_wait_tensorcnt pipelining); WMMA reads 128-bit
// LDS slices. Fallback: vectorized global loads (weights non-temporal).
// ---------------------------------------------------------------------------
__global__ __launch_bounds__(32) void fc1_wmma_kernel(
    const float* __restrict__ p,     // [16][Ktot]
    const float* __restrict__ fw,    // [128][Ktot]
    float* __restrict__ out1,        // [16][128] fp32 partials (zeroed)
    int Ktot, int steps)
{
    const int tid   = threadIdx.x;
    const int jcol  = tid & 15;
    const int g     = tid >> 4;
    const int ntile = blockIdx.x;
    const int k0    = blockIdx.y * steps * 32;
    const float* fwrow0 = fw + (size_t)(ntile * 16) * (size_t)Ktot;

    v8f acc = {};

#if defined(HAVE_TDM)
    __shared__ float ldsA[2][16][32];
    __shared__ float ldsB[2][16][32];

    // prologue: stage step 0 into buffer 0
    tdm_load_tile_16x32(p + k0,      lds_off(&ldsA[0][0][0]), (unsigned)Ktot);
    tdm_load_tile_16x32(fwrow0 + k0, lds_off(&ldsB[0][0][0]), (unsigned)Ktot);

    for (int st = 0; st < steps; ++st) {
        const int cbuf = st & 1;
        if (st + 1 < steps) {
            const int nbuf = (st + 1) & 1;
            const int kb   = k0 + (st + 1) * 32;
            tdm_load_tile_16x32(p + kb,      lds_off(&ldsA[nbuf][0][0]), (unsigned)Ktot);
            tdm_load_tile_16x32(fwrow0 + kb, lds_off(&ldsB[nbuf][0][0]), (unsigned)Ktot);
            __builtin_amdgcn_s_wait_tensorcnt(2);   // current buffer's 2 DMAs done
        } else {
            __builtin_amdgcn_s_wait_tensorcnt(0);
        }
        asm volatile("" ::: "memory");

        const float* Ar = &ldsA[cbuf][jcol][0];
        const float* Br = &ldsB[cbuf][jcol][0];
        v4f a0 = *(const v4f*)(Ar + g * 8);
        v4f a1 = *(const v4f*)(Ar + g * 8 + 4);
        v4f a2 = *(const v4f*)(Ar + 16 + g * 8);
        v4f a3 = *(const v4f*)(Ar + 16 + g * 8 + 4);
        v4f b0 = *(const v4f*)(Br + g * 8);
        v4f b1 = *(const v4f*)(Br + g * 8 + 4);
        v4f b2 = *(const v4f*)(Br + 16 + g * 8);
        v4f b3 = *(const v4f*)(Br + 16 + g * 8 + 4);

        v16bf a, b;
        pack8(a, 0, a0, a1); pack8(a, 8, a2, a3);
        pack8(b, 0, b0, b1); pack8(b, 8, b2, b3);
        acc = __builtin_amdgcn_wmma_f32_16x16x32_bf16(
            false, a, false, b, (short)0, acc, false, false);
    }
#else
    const float* Arow = p + (size_t)jcol * (size_t)Ktot;
    const float* Brow = fwrow0 + (size_t)jcol * (size_t)Ktot;
    for (int st = 0; st < steps; ++st) {
        const int kbase = k0 + st * 32;
        v4f a0 = *(const v4f*)(Arow + kbase + g * 8);
        v4f a1 = *(const v4f*)(Arow + kbase + g * 8 + 4);
        v4f a2 = *(const v4f*)(Arow + kbase + 16 + g * 8);
        v4f a3 = *(const v4f*)(Arow + kbase + 16 + g * 8 + 4);
        v4f b0 = __builtin_nontemporal_load((const v4f*)(Brow + kbase + g * 8));
        v4f b1 = __builtin_nontemporal_load((const v4f*)(Brow + kbase + g * 8 + 4));
        v4f b2 = __builtin_nontemporal_load((const v4f*)(Brow + kbase + 16 + g * 8));
        v4f b3 = __builtin_nontemporal_load((const v4f*)(Brow + kbase + 16 + g * 8 + 4));
        v16bf a, b;
        pack8(a, 0, a0, a1); pack8(a, 8, a2, a3);
        pack8(b, 0, b0, b1); pack8(b, 8, b2, b3);
        acc = __builtin_amdgcn_wmma_f32_16x16x32_bf16(
            false, a, false, b, (short)0, acc, false, false);
    }
#endif

    #pragma unroll
    for (int r = 0; r < 8; ++r) {
        int nrow = r + 8 * g;   // D: M = r + 8*lanegroup = batch row
        atomicAdd(&out1[nrow * 128 + ntile * 16 + jcol], acc[r]);
    }
}

// ---------------------------------------------------------------------------
// fc1 bias + fc2 + log_softmax; one lane per batch row (tiny)
// ---------------------------------------------------------------------------
__global__ __launch_bounds__(32) void head_kernel(
    const float* __restrict__ out1,   // [16][128]
    const float* __restrict__ fc1_b,  // [128]
    const float* __restrict__ fc2_w,  // [10][128]
    const float* __restrict__ fc2_b,  // [10]
    float* __restrict__ out)          // [16][10]
{
    int n = threadIdx.x;
    if (n >= NB) return;
    float h1[128];
    for (int k = 0; k < 128; ++k) h1[k] = out1[n * 128 + k] + fc1_b[k];
    float z[10];
    float m = -3.402823466e+38f;
    for (int j = 0; j < 10; ++j) {
        float s = fc2_b[j];
        for (int k = 0; k < 128; ++k) s += h1[k] * fc2_w[j * 128 + k];
        z[j] = s;
        m = fmaxf(m, s);
    }
    float se = 0.f;
    for (int j = 0; j < 10; ++j) se += expf(z[j] - m);
    float lse = m + logf(se);
    for (int j = 0; j < 10; ++j) out[n * 10 + j] = z[j] - lse;
}

// ---------------------------------------------------------------------------
extern "C" void kernel_launch(void* const* d_in, const int* in_sizes, int n_in,
                              void* d_out, int out_size, void* d_ws, size_t ws_size,
                              hipStream_t stream)
{
    const float* x     = (const float*)d_in[0];
    const float* depth = (const float*)d_in[1];
    const float* fx    = (const float*)d_in[2];
    const float* c1w0  = (const float*)d_in[3];
    const float* c1w1  = (const float*)d_in[4];
    const float* c1w2  = (const float*)d_in[5];
    const float* c1b   = (const float*)d_in[6];
    const float* c2w0  = (const float*)d_in[7];
    const float* c2w1  = (const float*)d_in[8];
    const float* c2w2  = (const float*)d_in[9];
    const float* c2b   = (const float*)d_in[10];
    const float* fc1w  = (const float*)d_in[11];
    const float* fc1b  = (const float*)d_in[12];
    const float* fc2w  = (const float*)d_in[13];
    const float* fc2b  = (const float*)d_in[14];

    const size_t ACT = (size_t)NB * OC * IMG_H * IMG_W;     // 12,845,056
    float* ws = (float*)d_ws;
    float* x1 = ws;            // conv1 out [16][16][224][224]
    float* x2 = ws + ACT;      // conv2 out (relu'd)
    float* p  = ws;            // pooled [16][16][112][112] — aliases x1 (dead)
    float* o1 = x2 + ACT;      // fc1 partials [16][128]

    dim3 cgrid(IMG_W / 16, IMG_H, NB);   // (14, 224, 16)

    conv25d_wmma_kernel<3, 1><<<cgrid, 32, 0, stream>>>(
        x, depth, fx, c1w0, c1w1, c1w2, c1b, x1, IMG_H, IMG_W);

    conv25d_wmma_kernel<16, 5><<<cgrid, 32, 0, stream>>>(
        x1, depth, fx, c2w0, c2w1, c2w2, c2b, x2, IMG_H, IMG_W);

    const int pool_total = NB * OC * 112 * 112;              // 3,211,264
    maxpool2_kernel<<<(pool_total + 255) / 256, 256, 0, stream>>>(x2, p, pool_total);

    hipMemsetAsync(o1, 0, NB * 128 * sizeof(float), stream);

    const int Ktot = OC * 112 * 112;                         // 200,704
    // 6272 WMMA K-steps total; split-K over 112 blocks => 56 steps each
    fc1_wmma_kernel<<<dim3(8, 112), 32, 0, stream>>>(p, fc1w, o1, Ktot, 56);

    head_kernel<<<1, 32, 0, stream>>>(o1, fc1b, fc2w, fc2b, (float*)d_out);
}
